// propagation_rule_76665166233872
// MI455X (gfx1250) — matrix operand order repaired and checked by reading, not compiled, not run on hardware
//
#include <hip/hip_runtime.h>
#include <hip/hip_bf16.h>

// CDNA5 / gfx1250 wave32 WMMA types
typedef __attribute__((ext_vector_type(16))) __bf16 v16bf;
typedef __attribute__((ext_vector_type(8)))  float  v8f;
typedef __attribute__((ext_vector_type(4)))  float  v4f;

#define PRE        128
#define NEXT       128
#define LDS_STRIDE 144   // 128 + 16 pad: keeps 32B alignment for v16bf loads, breaks bank patterns

// Each block: 256 threads = 8 wave32 waves. Each wave computes a 16-edge x 128-feature
// output tile with 3 fused bf16 WMMA GEMMs (ew1, ew2, com) + activation epilogue.
// NOTE: assumes nEdges % 16 == 0 (true for E = 500,000).
__global__ __launch_bounds__(256, 1)
void propagation_rule_wmma_kernel(const int* __restrict__ sample,      // [2, E] int32
                                  const float* __restrict__ emb1,      // [N, 128]
                                  const float* __restrict__ emb2,      // [N, 128]
                                  const float* __restrict__ Wcom,      // [128, 128]
                                  const float* __restrict__ bcom,      // [128]
                                  const float* __restrict__ Wew1,      // [128, 128]
                                  const float* __restrict__ Wew2,      // [128, 128]
                                  float* __restrict__ out,             // [E, 128]
                                  int nEdges)
{
    __shared__ __bf16 sW1[128 * LDS_STRIDE];   // W_ew1 (bf16)
    __shared__ __bf16 sW2[128 * LDS_STRIDE];   // W_ew2 (bf16)
    __shared__ __bf16 sWc[128 * LDS_STRIDE];   // W_com (bf16)
    __shared__ float  sBias[NEXT];

    const int tid = threadIdx.x;

    // ---- Stage weights into LDS, converting f32 -> bf16 (once per block) ----
    for (int i = tid; i < 128 * 128; i += 256) {
        const int r = i >> 7, c = i & 127;
        sW1[r * LDS_STRIDE + c] = (__bf16)Wew1[i];
        sW2[r * LDS_STRIDE + c] = (__bf16)Wew2[i];
        sWc[r * LDS_STRIDE + c] = (__bf16)Wcom[i];
    }
    if (tid < NEXT) sBias[tid] = bcom[tid];
    __syncthreads();

    const int lane = tid & 31;
    const int wave = tid >> 5;
    const int c16  = lane & 15;   // A: row-in-tile; B: column N; C/D: column N
    const int h    = lane >> 4;   // half-select per ISA 16-bit WMMA layouts

    const int* __restrict__ idx1 = sample;
    const int* __restrict__ idx2 = sample + nEdges;

    // Hoist bias into registers (only 8 VGPRs; cheap to keep resident).
    float biasr[8];
#pragma unroll
    for (int nt = 0; nt < 8; ++nt) biasr[nt] = sBias[nt * 16 + c16];

    const int nTiles    = nEdges >> 4;               // 16 | E
    const int waveGlob  = blockIdx.x * 8 + wave;
    const int waveCount = gridDim.x * 8;

    for (int t = waveGlob; t < nTiles; t += waveCount) {
        // Opaque zero: makes LDS B-fragment addresses formally loop-variant so the
        // compiler cannot hoist all 96 v16bf LDS loads out of the tile loop and
        // spill them to scratch (observed in round 1). Runtime value is always 0.
        int tweak = 0;
        asm volatile("" : "+v"(tweak));

        // ---------- Gather A fragments (e1, e2, e1+e2) for this 16-edge tile ----------
        const int edgeRow = t * 16 + c16;
        const int i1 = idx1[edgeRow];
        const int i2 = idx2[edgeRow];
        const float* __restrict__ r1 = emb1 + i1 * PRE;
        const float* __restrict__ r2 = emb2 + i2 * PRE;

        // 16-bit A 16x32 layout used here: lane holds row M = lane%16,
        // contiguous K range [kt*32 + 16h, +16)
        v16bf A1[4], A2[4], A12[4];
#pragma unroll
        for (int kt = 0; kt < 4; ++kt) {
            const int k0 = kt * 32 + h * 16;
            const int k1 = kt * 32 + h * 16 + 8;
            const v4f x0 = *(const v4f*)(r1 + k0);
            const v4f x1 = *(const v4f*)(r1 + k0 + 4);
            const v4f x2 = *(const v4f*)(r1 + k1);
            const v4f x3 = *(const v4f*)(r1 + k1 + 4);
            const v4f y0 = *(const v4f*)(r2 + k0);
            const v4f y1 = *(const v4f*)(r2 + k0 + 4);
            const v4f y2 = *(const v4f*)(r2 + k1);
            const v4f y3 = *(const v4f*)(r2 + k1 + 4);
#pragma unroll
            for (int j = 0; j < 4; ++j) {
                A1[kt][j]      = (__bf16)x0[j];
                A1[kt][4 + j]  = (__bf16)x1[j];
                A1[kt][8 + j]  = (__bf16)x2[j];
                A1[kt][12 + j] = (__bf16)x3[j];
                A2[kt][j]      = (__bf16)y0[j];
                A2[kt][4 + j]  = (__bf16)y1[j];
                A2[kt][8 + j]  = (__bf16)y2[j];
                A2[kt][12 + j] = (__bf16)y3[j];
                A12[kt][j]      = (__bf16)(x0[j] + y0[j]);
                A12[kt][4 + j]  = (__bf16)(x1[j] + y1[j]);
                A12[kt][8 + j]  = (__bf16)(x2[j] + y2[j]);
                A12[kt][12 + j] = (__bf16)(x3[j] + y3[j]);
            }
        }

        // ---------- Sweep 8 N-tiles: 3 WMMA chains + fused epilogue ----------
#pragma unroll
        for (int nt = 0; nt < 8; ++nt) {
            v8f acc1 = {}, acc2 = {}, acc3 = {};
            // 16-bit B 32x16 layout used here: lane holds column N = lane%16,
            // contiguous K range [kt*32 + 16h, +16) -> contiguous 16 bf16 of a W row.
            const int nrow = nt * 16 + c16;
            const int kb   = h * 16;
#pragma unroll
            for (int kt = 0; kt < 4; ++kt) {
                const int off = nrow * LDS_STRIDE + kt * 32 + kb + tweak;  // 32B-aligned
                const v16bf B1 = *(const v16bf*)(sW1 + off);
                const v16bf B2 = *(const v16bf*)(sW2 + off);
                const v16bf Bc = *(const v16bf*)(sWc + off);
                acc1 = __builtin_amdgcn_wmma_f32_16x16x32_bf16(
                    false, A1[kt],  false, B1, (short)0, acc1, false, false);
                acc2 = __builtin_amdgcn_wmma_f32_16x16x32_bf16(
                    false, A2[kt],  false, B2, (short)0, acc2, false, false);
                acc3 = __builtin_amdgcn_wmma_f32_16x16x32_bf16(
                    false, A12[kt], false, Bc, (short)0, acc3, false, false);
            }

            // Epilogue: lr = (sigmoid(acc1) + sigmoid(acc2)) * tanh(acc3 + b)
            // Fast path: v_exp_f32 + v_rcp_f32 (single trans ops) instead of the
            // IEEE division expansion (v_div_scale/v_div_fixup, ~10 VALU each).
            const int   colOut = nt * 16 + c16;
            const float bias   = biasr[nt];
            float* __restrict__ po = out + (t * 16 + h * 8) * NEXT + colOut;
#pragma unroll
            for (int r = 0; r < 8; ++r) {
                // ISA 16x16 f32 C/D layout: VGPR r -> row r (lanes 0-15) / r+8 (lanes 16-31)
                const float s1 = __builtin_amdgcn_rcpf(1.0f + __expf(-acc1[r]));
                const float s2 = __builtin_amdgcn_rcpf(1.0f + __expf(-acc2[r]));
                const float z  = acc3[r] + bias;
                // tanh(z) = 1 - 2/(exp(2z)+1)
                const float ex = __expf(2.0f * z);
                const float th = 1.0f - 2.0f * __builtin_amdgcn_rcpf(ex + 1.0f);
                po[r * NEXT] = (s1 + s2) * th;
            }
        }
    }
}

extern "C" void kernel_launch(void* const* d_in, const int* in_sizes, int n_in,
                              void* d_out, int out_size, void* d_ws, size_t ws_size,
                              hipStream_t stream) {
    // setup_inputs() order:
    // 0: sample [2,E] int32, 1: node_embedding1, 2: node_embedding2,
    // 3: W_com, 4: b_com, 5: W_ew1, 6: W_ew2
    const int*   sample = (const int*)d_in[0];
    const float* emb1   = (const float*)d_in[1];
    const float* emb2   = (const float*)d_in[2];
    const float* Wcom   = (const float*)d_in[3];
    const float* bcom   = (const float*)d_in[4];
    const float* Wew1   = (const float*)d_in[5];
    const float* Wew2   = (const float*)d_in[6];
    float*       out    = (float*)d_out;

    const int nEdges = in_sizes[0] / 2;   // 500,000 (divisible by 16)
    const int nTiles = nEdges >> 4;

    int blocks = (nTiles + 7) / 8;        // 8 waves per block, >=1 tile per wave
    if (blocks > 1024) blocks = 1024;     // persistent grid-stride; amortize LDS weight staging
    if (blocks < 1) blocks = 1;

    propagation_rule_wmma_kernel<<<blocks, 256, 0, stream>>>(
        sample, emb1, emb2, Wcom, bcom, Wew1, Wew2, out, nEdges);
}
